// GraphMultisetTransformer_62191126446174
// MI455X (gfx1250) — compile-verified, matrix operand order checked by reading, not run on hardware
//
#include <hip/hip_runtime.h>

// ---------------------------------------------------------------------------
// Types for CDNA5 WMMA
// ---------------------------------------------------------------------------
typedef __attribute__((ext_vector_type(16))) __bf16 v16bf;
typedef __attribute__((ext_vector_type(8)))  __bf16 v8bf;
typedef __attribute__((ext_vector_type(8)))  float  v8f;
typedef __attribute__((ext_vector_type(4)))  unsigned int v4u;

// GCC-style int4 vector: pointee type expected by the async-LDS b128 builtin
typedef int v4i32 __attribute__((vector_size(16)));
typedef __attribute__((address_space(1))) v4i32* as1_v4i32_p;   // global
typedef __attribute__((address_space(3))) v4i32* as3_v4i32_p;   // LDS

#define NN    8192      // nodes
#define FIN   128
#define HD    128       // H
#define DD    384       // D = 3H
#define NSEED 10        // K1 PMA seeds
#define NT    4         // 16x16 accum tiles per wave (wave covers 64 cols)

// CDNA5 async global->LDS path (ASYNCcnt), with safe fallback
#if defined(__has_builtin)
#if __has_builtin(__builtin_amdgcn_global_load_async_to_lds_b128) && \
    __has_builtin(__builtin_amdgcn_s_wait_asynccnt)
#define USE_ASYNC_LDS 1
#endif
#endif
#ifndef USE_ASYNC_LDS
#define USE_ASYNC_LDS 0
#endif

__device__ __forceinline__ void stage16B(const __bf16* gsrc, __bf16* ldst) {
#if USE_ASYNC_LDS
    __builtin_amdgcn_global_load_async_to_lds_b128(
        (as1_v4i32_p)(void*)gsrc,
        (as3_v4i32_p)(void*)ldst, 0, 0);
#else
    *(v4u*)ldst = *(const v4u*)gsrc;
#endif
}

__device__ __forceinline__ void stage_wait() {
#if USE_ASYNC_LDS
    __builtin_amdgcn_s_wait_asynccnt(0);
#endif
}

// ---------------------------------------------------------------------------
// fp32 -> bf16 convert
// ---------------------------------------------------------------------------
__global__ void f32_to_bf16(const float* __restrict__ in, __bf16* __restrict__ out, long n) {
    long i = (long)blockIdx.x * blockDim.x + threadIdx.x;
    if (i < n) out[i] = (__bf16)in[i];
}

__global__ void zero_f32(float* __restrict__ p, int n) {
    int i = blockIdx.x * blockDim.x + threadIdx.x;
    if (i < n) p[i] = 0.f;
}

// ---------------------------------------------------------------------------
// Pack fp32 B[K,N] (row-major) into per-lane WMMA bf16 fragment layout.
// Tile (ntile, kt) covers B[kt*32 .. +31][ntile*16 .. +15], stored as
// 32 lanes x 16 bf16 contiguous.  Lane (c,hi): elems 0..7 = K kb..kb+7,
// elems 8..15 = K kb+16..kb+23, kb = kt*32 + hi*8, column = ntile*16+c.
// Mirrors the 16-bit A-matrix layout of ISA 7.12.2 with M<->N swapped.
// ---------------------------------------------------------------------------
__global__ void pack_b(const float* __restrict__ B, __bf16* __restrict__ Bp, int K, int N) {
    int ktiles = K >> 5;
    int tiles  = (N >> 4) * ktiles;
    int gid = blockIdx.x * blockDim.x + threadIdx.x;
    if (gid >= tiles * 32) return;
    int t = gid >> 5, lane = gid & 31;
    int ntile = t / ktiles, kt = t % ktiles;
    int c = lane & 15, hi = lane >> 4;
    __bf16* o = Bp + (size_t)t * 512 + lane * 16;
    int col = ntile * 16 + c;
    int kb  = (kt << 5) + hi * 8;
#pragma unroll
    for (int e = 0; e < 8; ++e) o[e]     = (__bf16)B[(size_t)(kb + e) * N + col];
#pragma unroll
    for (int e = 0; e < 8; ++e) o[8 + e] = (__bf16)B[(size_t)(kb + 16 + e) * N + col];
}

// ---------------------------------------------------------------------------
// WMMA GEMM: C[8192,N] = op(A[8192,K](bf16,row-major,lda) @ B(packed) + bias)
// block = 256 threads = 8 waves; each wave: one 16-row tile, NT=4 col tiles.
// grid = (8192/128, N/64).  A tile double-buffered in LDS, staged with
// GLOBAL_LOAD_ASYNC_TO_LDS_B128 (ASYNCcnt) when available: one barrier per
// k-step, next tile's DMA overlapped with current tile's 4 WMMAs.
// ---------------------------------------------------------------------------
__global__ __launch_bounds__(256) void wmma_gemm(
    const __bf16* __restrict__ A, int lda,
    const __bf16* __restrict__ Bp,
    int K, int N,
    const float* __restrict__ bias, int relu,
    float* __restrict__ outF, int ldf,
    __bf16* __restrict__ outH, int ldh)
{
    __shared__ __attribute__((aligned(16))) __bf16 As[2][128][40];  // 32 cols + 8 pad

    const int tid  = threadIdx.x;
    const int wave = tid >> 5, lane = tid & 31;
    const int r = lane & 15, hi = lane >> 4;
    const int mblock = blockIdx.x * 128;
    const int n0 = blockIdx.y * (NT * 16);
    const int ktiles = K >> 5;

    // staging chunk assignment: 512 x 16B chunks, 2 per thread
    const int c0row = tid >> 2,          c0q = tid & 3;
    const int c1row = (tid + 256) >> 2,  c1q = (tid + 256) & 3;
    const __bf16* Arow0 = A + (size_t)(mblock + c0row) * lda + c0q * 8;
    const __bf16* Arow1 = A + (size_t)(mblock + c1row) * lda + c1q * 8;

    v8f acc[NT];
#pragma unroll
    for (int i = 0; i < NT; ++i)
#pragma unroll
        for (int j = 0; j < 8; ++j) acc[i][j] = 0.f;

    // prologue: stage k-tile 0 into buffer 0
    stage16B(Arow0, &As[0][c0row][c0q * 8]);
    stage16B(Arow1, &As[0][c1row][c1q * 8]);

    for (int kt = 0; kt < ktiles; ++kt) {
        const int cur = kt & 1;
        stage_wait();              // my async writes to As[cur] done
        __syncthreads();           // everyone's writes done (dscnt waited pre-barrier)

        if (kt + 1 < ktiles) {     // overlap next tile's DMA with compute
            const int k1 = (kt + 1) << 5;
            stage16B(Arow0 + k1, &As[cur ^ 1][c0row][c0q * 8]);
            stage16B(Arow1 + k1, &As[cur ^ 1][c1row][c1q * 8]);
        }

        // A fragment (ISA 7.12.2 16-bit 16x32 layout)
        const int arow = wave * 16 + r;
        v8bf alo = *(const v8bf*)&As[cur][arow][hi * 8];
        v8bf ahi = *(const v8bf*)&As[cur][arow][16 + hi * 8];
        v16bf afrag = __builtin_shufflevector(alo, ahi,
            0,1,2,3,4,5,6,7,8,9,10,11,12,13,14,15);

#pragma unroll
        for (int nt = 0; nt < NT; ++nt) {
            size_t ntg = (size_t)((n0 >> 4) + nt);
            const __bf16* bp = Bp + (ntg * ktiles + kt) * 512 + lane * 16;
            v8bf blo = *(const v8bf*)bp;
            v8bf bhi = *(const v8bf*)(bp + 8);
            v16bf bfrag = __builtin_shufflevector(blo, bhi,
                0,1,2,3,4,5,6,7,8,9,10,11,12,13,14,15);
            acc[nt] = __builtin_amdgcn_wmma_f32_16x16x32_bf16(
                false, afrag, false, bfrag, (short)0, acc[nt], false, false);
        }
    }

    // store: C/D layout -> elem i of lane (r,hi) is row (i + 8*hi), col r
    const int m0 = mblock + wave * 16;
#pragma unroll
    for (int nt = 0; nt < NT; ++nt) {
        int col = n0 + nt * 16 + r;
#pragma unroll
        for (int i = 0; i < 8; ++i) {
            int m = m0 + i + 8 * hi;
            float v = acc[nt][i];
            if (bias) v += bias[col];
            if (relu) v = v > 0.f ? v : 0.f;
            if (outF) outF[(size_t)m * ldf + col] = v;
            if (outH) outH[(size_t)m * ldh + col] = (__bf16)v;
        }
    }
}

// ---------------------------------------------------------------------------
// Small fp32 matmul: C[M,N] = [residual +] op(A[M,K] @ B[K,N] + bias)
// ---------------------------------------------------------------------------
__global__ void smm(const float* __restrict__ A, const float* __restrict__ B,
                    const float* __restrict__ bias, const float* __restrict__ residual,
                    float* __restrict__ C, int M, int K, int N, int relu)
{
    int idx = blockIdx.x * blockDim.x + threadIdx.x;
    if (idx >= M * N) return;
    int m = idx / N, n = idx % N;
    float s = bias ? bias[n] : 0.f;
    const float* ar = A + (size_t)m * K;
    for (int k = 0; k < K; ++k) s += ar[k] * B[(size_t)k * N + n];
    if (relu) s = s > 0.f ? s : 0.f;
    if (residual) s += residual[idx];
    C[idx] = s;
}

// ---------------------------------------------------------------------------
// Row layernorm (keras eps=1e-3), optional elementwise addend
// ---------------------------------------------------------------------------
__global__ void layernorm_rows(const float* __restrict__ x, const float* __restrict__ addA,
                               const float* __restrict__ g, const float* __restrict__ b,
                               float* __restrict__ out, int cols)
{
    __shared__ float s1[128], s2[128];
    int row = blockIdx.x, tid = threadIdx.x;
    const float* xr = x + (size_t)row * cols;
    const float* ar = addA ? addA + (size_t)row * cols : nullptr;
    float a0 = 0.f, a1 = 0.f;
    for (int c = tid; c < cols; c += 128) {
        float v = xr[c] + (ar ? ar[c] : 0.f);
        a0 += v; a1 += v * v;
    }
    s1[tid] = a0; s2[tid] = a1;
    __syncthreads();
    for (int st = 64; st > 0; st >>= 1) {
        if (tid < st) { s1[tid] += s1[tid + st]; s2[tid] += s2[tid + st]; }
        __syncthreads();
    }
    float mean = s1[0] / cols;
    float var  = s2[0] / cols - mean * mean;
    float inv  = rsqrtf(var + 1e-3f);
    for (int c = tid; c < cols; c += 128) {
        float v = xr[c] + (ar ? ar[c] : 0.f);
        out[(size_t)row * cols + c] = (v - mean) * inv * g[c] + b[c];
    }
}

// ---------------------------------------------------------------------------
// PMA1 attention: softmax over QUERY axis per (head, key n); accumulate A@V_
// into OAV[10,384] (merged-head layout) with block partials + atomics.
// ---------------------------------------------------------------------------
__global__ __launch_bounds__(256) void pma1_attn(
    const float* __restrict__ Q1,          // [10,384]
    const float* __restrict__ Kf,          // [8192,384]
    const float* __restrict__ Vf,          // [8192,384]
    float* __restrict__ OAV)               // [10,384], pre-zeroed
{
    __shared__ float Qs[NSEED * DD];       // 15 KB
    __shared__ float Ws[256][20];          // 20 KB
    const int tid = threadIdx.x;
    for (int i = tid; i < NSEED * DD; i += 256) Qs[i] = Q1[i];
    __syncthreads();

    const int n = blockIdx.x * 256 + tid;
    const float scale = 0.05103103630798288f;   // 1/sqrt(384)
#pragma unroll
    for (int h = 0; h < 2; ++h) {
        float l[NSEED]; float mx = -1e30f;
        const float* kp = Kf + (size_t)n * DD + h * 192;
#pragma unroll
        for (int q = 0; q < NSEED; ++q) {
            const float* qp = Qs + q * DD + h * 192;
            float d = 0.f;
            for (int c = 0; c < 192; ++c) d += qp[c] * kp[c];
            l[q] = d * scale;
            mx = fmaxf(mx, l[q]);
        }
        float s = 0.f;
#pragma unroll
        for (int q = 0; q < NSEED; ++q) { l[q] = __expf(l[q] - mx); s += l[q]; }
        float inv = 1.f / s;
#pragma unroll
        for (int q = 0; q < NSEED; ++q) Ws[tid][h * 10 + q] = l[q] * inv;
    }
    __syncthreads();

    const int n0 = blockIdx.x * 256;
    for (int o = tid; o < NSEED * DD; o += 256) {
        int q = o / DD, col = o % DD;
        int h = (col >= 192) ? 1 : 0;
        float s = 0.f;
        for (int nn = 0; nn < 256; ++nn)
            s += Ws[nn][h * 10 + q] * Vf[(size_t)(n0 + nn) * DD + col];
        atomicAdd(&OAV[o], s);
    }
}

// ---------------------------------------------------------------------------
// Small MAB attention (single block): out = Q + merge(A @ V_),
// A = softmax over query axis of Q_ K_^T * scale.
// ---------------------------------------------------------------------------
__global__ void small_attn(const float* __restrict__ Q, const float* __restrict__ K,
                           const float* __restrict__ V, float* __restrict__ out,
                           int Mq, int Nk, int dm, int heads, float scale)
{
    __shared__ float L[512];
    const int dh = dm / heads;
    const int tid = threadIdx.x;
    const int tot = heads * Mq * Nk;
    for (int i = tid; i < tot; i += 256) {
        int h = i / (Mq * Nk), rem = i % (Mq * Nk);
        int q = rem / Nk, k = rem % Nk;
        float s = 0.f;
        for (int d = 0; d < dh; ++d) s += Q[q * dm + h * dh + d] * K[k * dm + h * dh + d];
        L[(h * Mq + q) * Nk + k] = s * scale;
    }
    __syncthreads();
    for (int i = tid; i < heads * Nk; i += 256) {
        int h = i / Nk, k = i % Nk;
        float mx = -1e30f;
        for (int q = 0; q < Mq; ++q) mx = fmaxf(mx, L[(h * Mq + q) * Nk + k]);
        float s = 0.f;
        for (int q = 0; q < Mq; ++q) {
            float e = __expf(L[(h * Mq + q) * Nk + k] - mx);
            L[(h * Mq + q) * Nk + k] = e; s += e;
        }
        float inv = 1.f / s;
        for (int q = 0; q < Mq; ++q) L[(h * Mq + q) * Nk + k] *= inv;
    }
    __syncthreads();
    for (int o = tid; o < Mq * dm; o += 256) {
        int q = o / dm, col = o % dm, h = col / dh;
        float s = 0.f;
        for (int k = 0; k < Nk; ++k) s += L[(h * Mq + q) * Nk + k] * V[k * dm + col];
        out[o] = Q[o] + s;
    }
}

__global__ void softmax_vec(const float* __restrict__ z, float* __restrict__ out, int n) {
    if (threadIdx.x == 0 && blockIdx.x == 0) {
        float mx = z[0];
        for (int i = 1; i < n; ++i) mx = fmaxf(mx, z[i]);
        float s = 0.f;
        for (int i = 0; i < n; ++i) s += __expf(z[i] - mx);
        for (int i = 0; i < n; ++i) out[i] = __expf(z[i] - mx) / s;
    }
}

// ---------------------------------------------------------------------------
// Host orchestration
// ---------------------------------------------------------------------------
extern "C" void kernel_launch(void* const* d_in, const int* in_sizes, int n_in,
                              void* d_out, int out_size, void* d_ws, size_t ws_size,
                              hipStream_t stream)
{
    (void)in_sizes; (void)n_in; (void)out_size; (void)ws_size;
    const float* x   = (const float*)d_in[0];
    const float* a   = (const float*)d_in[1];
    const float* cW1 = (const float*)d_in[2];  const float* cb1 = (const float*)d_in[3];
    const float* cW2 = (const float*)d_in[4];  const float* cb2 = (const float*)d_in[5];
    const float* cW3 = (const float*)d_in[6];  const float* cb3 = (const float*)d_in[7];
    const float* S1  = (const float*)d_in[8];
    const float* q1W = (const float*)d_in[9];  const float* q1b = (const float*)d_in[10];
    const float* k1W = (const float*)d_in[11]; const float* k1b = (const float*)d_in[12];
    const float* v1W = (const float*)d_in[13]; const float* v1b = (const float*)d_in[14];
    const float* o1W = (const float*)d_in[15]; const float* o1b = (const float*)d_in[16];
    const float* g1a = (const float*)d_in[17]; const float* b1a = (const float*)d_in[18];
    const float* g1b = (const float*)d_in[19]; const float* b1b = (const float*)d_in[20];
    const float* q2W = (const float*)d_in[21]; const float* q2b = (const float*)d_in[22];
    const float* k2W = (const float*)d_in[23]; const float* k2b = (const float*)d_in[24];
    const float* v2W = (const float*)d_in[25]; const float* v2b = (const float*)d_in[26];
    const float* o2W = (const float*)d_in[27]; const float* o2b = (const float*)d_in[28];
    const float* g2a = (const float*)d_in[29]; const float* b2a = (const float*)d_in[30];
    const float* g2b = (const float*)d_in[31]; const float* b2b = (const float*)d_in[32];
    const float* S3  = (const float*)d_in[33];
    const float* q3W = (const float*)d_in[34]; const float* q3b = (const float*)d_in[35];
    const float* k3W = (const float*)d_in[36]; const float* k3b = (const float*)d_in[37];
    const float* v3W = (const float*)d_in[38]; const float* v3b = (const float*)d_in[39];
    const float* o3W = (const float*)d_in[40]; const float* o3b = (const float*)d_in[41];
    const float* g3a = (const float*)d_in[42]; const float* b3a = (const float*)d_in[43];
    const float* g3b = (const float*)d_in[44]; const float* b3b = (const float*)d_in[45];
    const float* fW  = (const float*)d_in[46]; const float* fb  = (const float*)d_in[47];
    const float* c1W = (const float*)d_in[48]; const float* c1b = (const float*)d_in[49];
    const float* c2W = (const float*)d_in[50]; const float* c2b = (const float*)d_in[51];
    const float* c3W = (const float*)d_in[52]; const float* c3b = (const float*)d_in[53];

    // ---- workspace carve-out -------------------------------------------------
    char* ws = (char*)d_ws;
    size_t off = 0;
    auto take = [&](size_t bytes) -> char* {
        char* p = ws + off;
        off = (off + bytes + 255) & ~(size_t)255;
        return p;
    };
    __bf16* a_bf  = (__bf16*)take((size_t)NN * NN * 2);     // 128 MB (L2-resident)
    __bf16* x_bf  = (__bf16*)take((size_t)NN * FIN * 2);
    __bf16* xc_bf = (__bf16*)take((size_t)NN * DD * 2);     // [h1|h2|h3] bf16
    float*  t_f32 = (float*) take((size_t)NN * DD * 4);     // projection scratch
    __bf16* Bp    = (__bf16*)take((size_t)NN * DD * 2);     // packed B fragments
    float*  Kf    = (float*) take((size_t)NN * DD * 4);
    float*  Vf    = (float*) take((size_t)NN * DD * 4);
    float*  Q1    = (float*) take(NSEED * DD * 4);
    float*  OAV   = (float*) take(NSEED * DD * 4);
    float*  T1    = (float*) take(NSEED * DD * 4);
    float*  T2    = (float*) take(NSEED * DD * 4);
    float*  T3    = (float*) take(NSEED * DD * 4);
    float*  T4    = (float*) take(NSEED * DD * 4);
    float*  T5    = (float*) take(NSEED * DD * 4);
    float*  T6    = (float*) take(NSEED * DD * 4);
    float*  zbuf  = (float*) take(4 * 512);

    const dim3 blk(256);
    auto cdiv = [](long n, long d) { return (unsigned)((n + d - 1) / d); };

    // ---- precision conversion ----------------------------------------------
    f32_to_bf16<<<cdiv((long)NN * NN, 256), blk, 0, stream>>>(a, a_bf, (long)NN * NN);
    f32_to_bf16<<<cdiv((long)NN * FIN, 256), blk, 0, stream>>>(x, x_bf, (long)NN * FIN);

    // ---- GCN conv 1: h1 = relu(a @ (x@cW1) + cb1) --------------------------
    pack_b<<<cdiv((FIN/16)*(FIN/32)*32, 256), blk, 0, stream>>>(cW1, Bp, FIN, HD);
    wmma_gemm<<<dim3(NN/128, HD/64), blk, 0, stream>>>(x_bf, FIN, Bp, FIN, HD,
        nullptr, 0, t_f32, HD, nullptr, 0);
    pack_b<<<cdiv((HD/16)*(NN/32)*32, 256), blk, 0, stream>>>(t_f32, Bp, NN, HD);
    wmma_gemm<<<dim3(NN/128, HD/64), blk, 0, stream>>>(a_bf, NN, Bp, NN, HD,
        cb1, 1, nullptr, 0, xc_bf + 0, DD);

    // ---- GCN conv 2 --------------------------------------------------------
    pack_b<<<cdiv((HD/16)*(HD/32)*32, 256), blk, 0, stream>>>(cW2, Bp, HD, HD);
    wmma_gemm<<<dim3(NN/128, HD/64), blk, 0, stream>>>(xc_bf + 0, DD, Bp, HD, HD,
        nullptr, 0, t_f32, HD, nullptr, 0);
    pack_b<<<cdiv((HD/16)*(NN/32)*32, 256), blk, 0, stream>>>(t_f32, Bp, NN, HD);
    wmma_gemm<<<dim3(NN/128, HD/64), blk, 0, stream>>>(a_bf, NN, Bp, NN, HD,
        cb2, 1, nullptr, 0, xc_bf + 128, DD);

    // ---- GCN conv 3 --------------------------------------------------------
    pack_b<<<cdiv((HD/16)*(HD/32)*32, 256), blk, 0, stream>>>(cW3, Bp, HD, HD);
    wmma_gemm<<<dim3(NN/128, HD/64), blk, 0, stream>>>(xc_bf + 128, DD, Bp, HD, HD,
        nullptr, 0, t_f32, HD, nullptr, 0);
    pack_b<<<cdiv((HD/16)*(NN/32)*32, 256), blk, 0, stream>>>(t_f32, Bp, NN, HD);
    wmma_gemm<<<dim3(NN/128, HD/64), blk, 0, stream>>>(a_bf, NN, Bp, NN, HD,
        cb3, 1, nullptr, 0, xc_bf + 256, DD);

    // ---- PMA1 K = a @ (xc@k1W) + k1b ---------------------------------------
    pack_b<<<cdiv((DD/16)*(DD/32)*32, 256), blk, 0, stream>>>(k1W, Bp, DD, DD);
    wmma_gemm<<<dim3(NN/128, DD/64), blk, 0, stream>>>(xc_bf, DD, Bp, DD, DD,
        nullptr, 0, t_f32, DD, nullptr, 0);
    pack_b<<<cdiv((DD/16)*(NN/32)*32, 256), blk, 0, stream>>>(t_f32, Bp, NN, DD);
    wmma_gemm<<<dim3(NN/128, DD/64), blk, 0, stream>>>(a_bf, NN, Bp, NN, DD,
        k1b, 0, Kf, DD, nullptr, 0);

    // ---- PMA1 V = a @ (xc@v1W) + v1b ---------------------------------------
    pack_b<<<cdiv((DD/16)*(DD/32)*32, 256), blk, 0, stream>>>(v1W, Bp, DD, DD);
    wmma_gemm<<<dim3(NN/128, DD/64), blk, 0, stream>>>(xc_bf, DD, Bp, DD, DD,
        nullptr, 0, t_f32, DD, nullptr, 0);
    pack_b<<<cdiv((DD/16)*(NN/32)*32, 256), blk, 0, stream>>>(t_f32, Bp, NN, DD);
    wmma_gemm<<<dim3(NN/128, DD/64), blk, 0, stream>>>(a_bf, NN, Bp, NN, DD,
        v1b, 0, Vf, DD, nullptr, 0);

    // ---- PMA1 attention (softmax over query axis) --------------------------
    smm<<<cdiv(NSEED * DD, 256), blk, 0, stream>>>(S1, q1W, q1b, nullptr, Q1,
        NSEED, DD, DD, 0);
    zero_f32<<<cdiv(NSEED * DD, 256), blk, 0, stream>>>(OAV, NSEED * DD);
    pma1_attn<<<NN / 256, blk, 0, stream>>>(Q1, Kf, Vf, OAV);
    // O = LN(Q + A@V) ; O = LN(O + relu(O@o1W + o1b))
    layernorm_rows<<<NSEED, 128, 0, stream>>>(OAV, Q1, g1a, b1a, T1, DD);
    smm<<<cdiv(NSEED * DD, 256), blk, 0, stream>>>(T1, o1W, o1b, T1, T2,
        NSEED, DD, DD, 1);
    layernorm_rows<<<NSEED, 128, 0, stream>>>(T2, nullptr, g1b, b1b, T1, DD); // bx1 in T1

    // ---- SAB: MAB(bx1, bx1) with dim_V = 128 -------------------------------
    smm<<<cdiv(NSEED * HD, 256), blk, 0, stream>>>(T1, q2W, q2b, nullptr, T2, NSEED, DD, HD, 0);
    smm<<<cdiv(NSEED * HD, 256), blk, 0, stream>>>(T1, k2W, k2b, nullptr, T3, NSEED, DD, HD, 0);
    smm<<<cdiv(NSEED * HD, 256), blk, 0, stream>>>(T1, v2W, v2b, nullptr, T4, NSEED, DD, HD, 0);
    small_attn<<<1, blk, 0, stream>>>(T2, T3, T4, T5, NSEED, NSEED, HD, 2,
        0.08838834764831845f);                                  // 1/sqrt(128)
    layernorm_rows<<<NSEED, 128, 0, stream>>>(T5, nullptr, g2a, b2a, T2, HD);
    smm<<<cdiv(NSEED * HD, 256), blk, 0, stream>>>(T2, o2W, o2b, T2, T3, NSEED, HD, HD, 1);
    layernorm_rows<<<NSEED, 128, 0, stream>>>(T3, nullptr, g2b, b2b, T2, HD); // bx2 in T2

    // ---- PMA2: MAB(S3, bx2), dim_V = 384, 1 seed ---------------------------
    smm<<<cdiv(DD, 256), blk, 0, stream>>>(S3, q3W, q3b, nullptr, T3, 1, DD, DD, 0);
    smm<<<cdiv(NSEED * DD, 256), blk, 0, stream>>>(T2, k3W, k3b, nullptr, T4, NSEED, HD, DD, 0);
    smm<<<cdiv(NSEED * DD, 256), blk, 0, stream>>>(T2, v3W, v3b, nullptr, T5, NSEED, HD, DD, 0);
    small_attn<<<1, blk, 0, stream>>>(T3, T4, T5, T6, 1, NSEED, DD, 2,
        0.05103103630798288f);                                  // 1/sqrt(384)
    layernorm_rows<<<1, 128, 0, stream>>>(T6, nullptr, g3a, b3a, T3, DD);
    smm<<<cdiv(DD, 256), blk, 0, stream>>>(T3, o3W, o3b, T3, T4, 1, DD, DD, 1);
    layernorm_rows<<<1, 128, 0, stream>>>(T4, nullptr, g3b, b3b, T3, DD);    // bx3 in T3

    // ---- head: Dense(F_IN) + classifier + softmax --------------------------
    float* z0 = zbuf, *z1 = zbuf + 128, *z2 = zbuf + 256, *z3 = zbuf + 384;
    smm<<<1, blk, 0, stream>>>(T3, fW,  fb,  nullptr, z0, 1, DD,  FIN, 0);
    smm<<<1, blk, 0, stream>>>(z0, c1W, c1b, nullptr, z1, 1, FIN, HD,  1);
    smm<<<1, blk, 0, stream>>>(z1, c2W, c2b, nullptr, z2, 1, HD,  HD/2, 1);
    smm<<<1, blk, 0, stream>>>(z2, c3W, c3b, nullptr, z3, 1, HD/2, 2,  0);
    softmax_vec<<<1, 64, 0, stream>>>(z3, (float*)d_out, 2);
}
